// arw_encoder_44985487458903
// MI455X (gfx1250) — matrix operand order repaired and checked by reading, not compiled, not run on hardware
//
#include <hip/hip_runtime.h>
#include <hip/hip_bf16.h>

typedef __attribute__((ext_vector_type(16))) _Float16 v16h;
typedef __attribute__((ext_vector_type(8)))  float    v8f;

#define NEG_SLOPE 0.01f

__device__ __forceinline__ float leaky(float v) {
    // slope < 1  =>  leaky(v) == max(v, slope*v), branchless
    return fmaxf(v, NEG_SLOPE * v);
}

// Sign-split float atomic max (correct for mixed signs when buffer init = -inf)
__device__ __forceinline__ void atomicMaxF(float* addr, float val) {
    if (val >= 0.0f) {
        atomicMax((int*)addr, __float_as_int(val));
    } else {
        atomicMin((unsigned int*)addr, __float_as_uint(val));
    }
}

__device__ __forceinline__ void chan9(const float* __restrict__ xp, float v[9]) {
    float a0 = xp[0], a1 = xp[1], a2 = xp[2];
    v[0] = a0; v[1] = a1; v[2] = a2;
    v[3] = a0 * a0; v[4] = a0 * a1; v[5] = a0 * a2;
    v[6] = a1 * a1; v[7] = a1 * a2; v[8] = a2 * a2;
}

// ---------------------------------------------------------------------------
// Kernel A: per-256-point-tile partial sums of 9 channels (x,y,z + outer prods)
// Wave32 butterfly reduce (no barriers) + one cross-wave combine barrier.
// ---------------------------------------------------------------------------
__global__ void k_tilesum(const float* __restrict__ x, float* __restrict__ ts,
                          int N, int T) {
    int blk = blockIdx.x;
    int b = blk / T, t = blk % T;
    int tid = threadIdx.x, lane = tid & 31, w = tid >> 5;
    int i = t * 256 + tid;
    float v[9];
    if (i < N) {
        chan9(x + ((size_t)b * N + i) * 3, v);
    } else {
        #pragma unroll
        for (int c = 0; c < 9; ++c) v[c] = 0.0f;
    }
    #pragma unroll
    for (int c = 0; c < 9; ++c) {
        float s = v[c];
        #pragma unroll
        for (int off = 16; off > 0; off >>= 1) s += __shfl_xor(s, off, 32);
        v[c] = s;
    }
    __shared__ float part[8][9];
    if (lane == 0) {
        #pragma unroll
        for (int c = 0; c < 9; ++c) part[w][c] = v[c];
    }
    __syncthreads();
    if (tid < 9) {
        float s = 0.0f;
        #pragma unroll
        for (int ww = 0; ww < 8; ++ww) s += part[ww][tid];
        ts[((size_t)(b * T + t)) * 9 + tid] = s;
    }
}

// ---------------------------------------------------------------------------
// Kernel B: exclusive scan across tiles, per batch (T ~= 391, trivial)
// ---------------------------------------------------------------------------
__global__ void k_scantiles(float* __restrict__ ts, int T) {
    int b = blockIdx.x;
    if (threadIdx.x == 0) {
        float run[9];
        #pragma unroll
        for (int c = 0; c < 9; ++c) run[c] = 0.0f;
        for (int t = 0; t < T; ++t) {
            float* p = ts + ((size_t)(b * T + t)) * 9;
            #pragma unroll
            for (int c = 0; c < 9; ++c) {
                float tmp = p[c];
                p[c] = run[c];
                run[c] += tmp;
            }
        }
    }
}

// ---------------------------------------------------------------------------
// Kernel C: full prefix P[b][i][c] via wave32 __shfl_up scan (no barriers)
// + single cross-wave combine barrier. P[b][0][c]=0; P[b][i+1][c]=incl. sum.
// ---------------------------------------------------------------------------
__global__ void k_prefix(const float* __restrict__ x, const float* __restrict__ ts,
                         float* __restrict__ P, int N, int T) {
    int blk = blockIdx.x;
    int b = blk / T, t = blk % T;
    int tid = threadIdx.x, lane = tid & 31, w = tid >> 5;
    int i = t * 256 + tid;
    float v[9];
    if (i < N) {
        chan9(x + ((size_t)b * N + i) * 3, v);
    } else {
        #pragma unroll
        for (int c = 0; c < 9; ++c) v[c] = 0.0f;
    }
    // wave-level inclusive scan, all 9 channels, zero barriers
    #pragma unroll
    for (int c = 0; c < 9; ++c) {
        float s = v[c];
        #pragma unroll
        for (int off = 1; off < 32; off <<= 1) {
            float tmp = __shfl_up(s, off, 32);
            if (lane >= off) s += tmp;
        }
        v[c] = s;
    }
    __shared__ float wsum[8][9];
    if (lane == 31) {
        #pragma unroll
        for (int c = 0; c < 9; ++c) wsum[w][c] = v[c];
    }
    __syncthreads();

    float* Pb = P + (size_t)b * (N + 1) * 9;
    if (t == 0 && tid < 9) Pb[tid] = 0.0f;

    if (i < N) {
        // offset = exclusive tile prefix + preceding-wave totals (w uniform/wave)
        const float* ep = ts + ((size_t)(b * T + t)) * 9;
        float off9[9];
        #pragma unroll
        for (int c = 0; c < 9; ++c) off9[c] = ep[c];
        for (int ww = 0; ww < w; ++ww) {
            #pragma unroll
            for (int c = 0; c < 9; ++c) off9[c] += wsum[ww][c];
        }
        float* dst = Pb + (size_t)(i + 1) * 9;
        #pragma unroll
        for (int c = 0; c < 9; ++c) dst[c] = v[c] + off9[c];
    }
}

// ---------------------------------------------------------------------------
// Kernel: init pooled-max buffer to -inf (must run every launch)
// ---------------------------------------------------------------------------
__global__ void k_initpool(float* __restrict__ pooled, int n) {
    int i = blockIdx.x * blockDim.x + threadIdx.x;
    if (i < n) pooled[i] = -__builtin_inff();
}

// ---------------------------------------------------------------------------
// Kernel D (main): window covariance -> 12-dim feature -> WMMA 12->128 Linear
//   -> raw-z max pool (leaky+bias deferred via monotonicity) -> atomic max.
//
// Block = 256 threads (8 waves), 512 points. Each wave owns 4 M-tiles
// (64 points) held as 4 A-fragments in registers; per N-tile the B fragment
// is loaded once from LDS and reused by 4 back-to-back WMMAs.
//
// LDS fragment staging: for both A (features) and B (W1), each 32-half row is
// laid out so that halves [0..15] are exactly the lane0-15 fragment and halves
// [16..31] are exactly the lane16-31 fragment (zeros pre-placed where the
// 16x16x32 K-layout requires them). Each lane does one unconditional aligned
// 32-byte LDS read -> 2x ds_load_b128, zero divergence.
// ---------------------------------------------------------------------------
__global__ __launch_bounds__(256) void
k_main(const float* __restrict__ x, const float* __restrict__ P,
       const float* __restrict__ W1, const float* __restrict__ b1,
       float* __restrict__ pooled, int N, int nT, int half, float inv_ns1) {
    int blk = blockIdx.x;
    int b = blk / nT, tile = blk % nT;
    int base = tile * 512;
    int tid  = threadIdx.x;        // 0..255
    int lane = tid & 31;
    int w    = tid >> 5;           // 0..7
    int hsel = lane & 16;          // 0 for lanes 0-15, 16 for lanes 16-31

    __shared__ __align__(32) _Float16 feath[512][32];
    __shared__ __align__(32) _Float16 w1x[128][32];
    __shared__ float wavemax[8][128];

    // Stage W1 (B-matrix layout): row j halves [0..7]=W1[j][0..7], [16..19]=W1[j][8..11], rest 0.
    // 256 threads: tid<128 fill low halves, tid>=128 fill high halves.
    {
        int j  = tid & 127;
        int hs = tid >> 7;                  // 0 or 1
        #pragma unroll
        for (int k = 0; k < 16; ++k) {
            _Float16 hv = (_Float16)0.0f;
            if (hs == 0 && k < 8) hv = (_Float16)W1[j * 12 + k];
            if (hs == 1 && k < 4) hv = (_Float16)W1[j * 12 + 8 + k];
            w1x[j][hs * 16 + k] = hv;
        }
    }

    // Features: each thread builds rows tid and tid+256. Clamped index:
    // max-pool is invariant under duplicating the last point -> no masking.
    #pragma unroll
    for (int rr = 0; rr < 2; ++rr) {
        int row = tid + rr * 256;
        int i = base + row; if (i >= N) i = N - 1;
        int st = i - half;     if (st < 0) st = 0;
        int en = i + half + 1; if (en > N) en = N;
        float cnt = (float)(en - st);
        const float* Pb = P + (size_t)b * (N + 1) * 9;
        const float* pe = Pb + (size_t)en * 9;
        const float* ps = Pb + (size_t)st * 9;
        float S[9];
        #pragma unroll
        for (int c = 0; c < 9; ++c) S[c] = pe[c] - ps[c];
        float ic = 1.0f / cnt;
        float m0 = S[0] * ic, m1 = S[1] * ic, m2 = S[2] * ic;
        float c00 = (S[3] - cnt * m0 * m0) * inv_ns1;
        float c01 = (S[4] - cnt * m0 * m1) * inv_ns1;
        float c02 = (S[5] - cnt * m0 * m2) * inv_ns1;
        float c11 = (S[6] - cnt * m1 * m1) * inv_ns1;
        float c12 = (S[7] - cnt * m1 * m2) * inv_ns1;
        float c22 = (S[8] - cnt * m2 * m2) * inv_ns1;
        const float* xp = x + ((size_t)b * N + i) * 3;
        float f[12] = { xp[0], xp[1], xp[2],
                        c00, c01, c02,
                        c01, c11, c12,
                        c02, c12, c22 };
        // A-matrix 16x32 f16 layout: lanes0-15 carry K0..7 (+K16..23=0),
        // lanes16-31 carry K8..15 (+K24..31=0).
        #pragma unroll
        for (int k = 0; k < 8; ++k)  feath[row][k]      = (_Float16)f[k];       // K0..7
        #pragma unroll
        for (int k = 8; k < 16; ++k) feath[row][k]      = (_Float16)0.0f;       // K16..23
        #pragma unroll
        for (int k = 0; k < 4; ++k)  feath[row][16 + k] = (_Float16)f[8 + k];   // K8..11
        #pragma unroll
        for (int k = 4; k < 16; ++k) feath[row][16 + k] = (_Float16)0.0f;       // K12..15,K24..31
    }
    __syncthreads();

    // 4 A-fragments per wave (64 points)
    v16h a[4];
    #pragma unroll
    for (int mt = 0; mt < 4; ++mt) {
        int mrow = w * 64 + mt * 16 + (lane & 15);
        a[mt] = *reinterpret_cast<const v16h*>(&feath[mrow][hsel]);
    }

    int colbase = lane & 15;
    #pragma unroll
    for (int n = 0; n < 8; ++n) {
        int col = n * 16 + colbase;
        v16h bf = *reinterpret_cast<const v16h*>(&w1x[col][hsel]);
        float m = -__builtin_inff();
        #pragma unroll
        for (int mt = 0; mt < 4; ++mt) {
            v8f c = {};
            c = __builtin_amdgcn_wmma_f32_16x16x32_f16(
                /*neg_a=*/false, a[mt], /*neg_b=*/false, bf,
                /*c_mod=*/(short)0, c, /*reuse_a=*/false, /*reuse_b=*/false);
            #pragma unroll
            for (int r = 0; r < 8; ++r) m = fmaxf(m, c[r]);
        }
        // combine complementary row-halves (same column)
        m = fmaxf(m, __shfl_xor(m, 16, 32));
        wavemax[w][col] = m;   // both halves store the identical value
    }
    __syncthreads();

    // Block combine; bias + leaky applied once (monotone, commutes with max)
    if (tid < 128) {
        float mz = wavemax[0][tid];
        #pragma unroll
        for (int ww = 1; ww < 8; ++ww) mz = fmaxf(mz, wavemax[ww][tid]);
        float v = leaky(mz + b1[tid]);
        atomicMaxF(pooled + b * 128 + tid, v);
    }
}

// ---------------------------------------------------------------------------
// Kernel E: out[b][j] = leaky(pooled[b] . W2[j] + b2[j]); 8x64 outputs
// ---------------------------------------------------------------------------
__global__ void k_final(const float* __restrict__ pooled, const float* __restrict__ W2,
                        const float* __restrict__ b2, float* __restrict__ out) {
    int tid = threadIdx.x;           // 512 threads
    int b = tid >> 6, j = tid & 63;
    const float* p  = pooled + b * 128;
    const float* wr = W2 + j * 128;
    float acc = b2[j];
    #pragma unroll 8
    for (int k = 0; k < 128; ++k) acc += p[k] * wr[k];
    out[b * 64 + j] = leaky(acc);
}

extern "C" void kernel_launch(void* const* d_in, const int* in_sizes, int n_in,
                              void* d_out, int out_size, void* d_ws, size_t ws_size,
                              hipStream_t stream) {
    const float* x  = (const float*)d_in[0];
    const float* W1 = (const float*)d_in[1];
    const float* b1 = (const float*)d_in[2];
    const float* W2 = (const float*)d_in[3];
    const float* b2 = (const float*)d_in[4];
    float* out = (float*)d_out;

    const int B = 8;
    const int N = 100000;                 // in_sizes[0] == B*N*3
    int ns = N / 50; if (ns < 10) ns = 10;
    int half = ns / 2;
    float inv_ns1 = 1.0f / (float)(ns - 1);

    const int T     = (N + 255) / 256;    // scan tiles per batch
    const int nT512 = (N + 511) / 512;    // main-kernel point tiles per batch

    // Workspace layout (floats): P[B][N+1][9] | tileSums[B][T][9] | pooled[B][128]
    float* ws = (float*)d_ws;
    size_t P_elems  = (size_t)B * (size_t)(N + 1) * 9;
    size_t ts_elems = (size_t)B * (size_t)T * 9;
    float* P      = ws;
    float* ts     = ws + P_elems;
    float* pooled = ts + ts_elems;

    k_tilesum  <<<B * T,     256, 0, stream>>>(x, ts, N, T);
    k_scantiles<<<B,          32, 0, stream>>>(ts, T);
    k_prefix   <<<B * T,     256, 0, stream>>>(x, ts, P, N, T);
    k_initpool <<<(B * 128 + 255) / 256, 256, 0, stream>>>(pooled, B * 128);
    k_main     <<<B * nT512, 256, 0, stream>>>(x, P, W1, b1, pooled, N, nT512, half, inv_ns1);
    k_final    <<<1,         512, 0, stream>>>(pooled, W2, b2, out);
}
